// KeywordSpottingNet_Spintronic_27556510171448
// MI455X (gfx1250) — matrix operand order repaired and checked by reading, not compiled, not run on hardware
//
#include <hip/hip_runtime.h>
#include <hip/hip_bf16.h>

// ---------------------------------------------------------------------------
// KeywordSpottingNet on MI455X (gfx1250): bf16 WMMA GEMMs + fused feature net.
// ---------------------------------------------------------------------------

typedef __attribute__((ext_vector_type(16))) __bf16        v16bf;
typedef __attribute__((ext_vector_type(8)))  float         v8f;
typedef __attribute__((ext_vector_type(4)))  unsigned int  u32x4;
typedef int async_v4i __attribute__((__vector_size__(16))); // matches builtin param

#define B_   512
#define S_   128
#define F_   40
#define H_   512
#define K_   10
#define FPAD 64
#define BS   (B_ * S_)   // 65536

// ----------------------------- helpers -------------------------------------

__device__ __forceinline__ unsigned short f2bf(float f) {
  unsigned int u = __float_as_uint(f);
  unsigned int r = (u + 0x7FFFu + ((u >> 16) & 1u)) >> 16;  // RNE
  return (unsigned short)r;
}
__device__ __forceinline__ float bf2f(unsigned short h) {
  return __uint_as_float(((unsigned int)h) << 16);
}
__device__ __forceinline__ float sigmoidf_(float x) {
  return 1.f / (1.f + __expf(-x));
}

// A fragment (16x32 bf16, M x K): lane m<16 holds row m, K = k0..k0+15;
// lane m+16 holds K = k0+16..k0+31 (same k order as load_fragB).
__device__ __forceinline__ v16bf load_fragA(const unsigned short* base, int ld,
                                            int row0, int k0, int lane) {
  int row = row0 + (lane & 15);
  int kb  = k0 + ((lane >> 4) << 4);
  const unsigned short* p = base + (long)row * ld + kb;
  union { u32x4 q[2]; v16bf v; } u;
  u.q[0] = *(const u32x4*)(p);
  u.q[1] = *(const u32x4*)(p + 8);
  return u.v;
}

// B fragment (32x16 bf16, K x N) built from row-major weight Wn[N][K]:
// lane n<16 holds W row n, K = k0..k0+15; lane n+16 holds K = k0+16..k0+31.
__device__ __forceinline__ v16bf load_fragB(const unsigned short* base, int ld,
                                            int col0, int k0, int lane) {
  int row = col0 + (lane & 15);
  int kb  = k0 + ((lane >> 4) << 4);
  const unsigned short* p = base + (long)row * ld + kb;
  union { u32x4 q[2]; v16bf v; } u;
  u.q[0] = *(const u32x4*)(p);
  u.q[1] = *(const u32x4*)(p + 8);
  return u.v;
}

#define WMMA_BF16(a, b, c) \
  __builtin_amdgcn_wmma_f32_16x16x32_bf16(false, (a), false, (b), (short)0, (c), false, false)

// ----------------------------- small kernels --------------------------------

__global__ void zero_u32_kernel(unsigned int* p, long n) {
  long i = (long)blockIdx.x * blockDim.x + threadIdx.x;
  if (i < n) p[i] = 0u;
}

// active_any[s] = any_b( sum_f x[b,s,f]^2 > 0.01 )
__global__ void energy_kernel(const float* __restrict__ x, int* __restrict__ flags) {
  int s = blockIdx.x;
  int any = 0;
  for (int b = threadIdx.x; b < B_; b += blockDim.x) {
    const float* xp = x + (size_t)b * S_ * F_ + (size_t)s * F_;
    float e = 0.f;
#pragma unroll
    for (int f = 0; f < F_; ++f) e += xp[f] * xp[f];
    if (e > 0.01f) any = 1;
  }
  if (any) atomicOr(&flags[s], 1);
}

// f32 [rows,kin] -> bf16 [rows,kout] with zero padding of K
__global__ void convert_pad_kernel(const float* __restrict__ src,
                                   unsigned short* __restrict__ dst,
                                   int rows, int kin, int kout) {
  long i = (long)blockIdx.x * blockDim.x + threadIdx.x;
  if (i >= (long)rows * kout) return;
  int r = (int)(i / kout), c = (int)(i % kout);
  float v = (c < kin) ? src[(long)r * kin + c] : 0.f;
  dst[i] = f2bf(v);
}

// ----------------------------- generic WMMA GEMM ----------------------------
// C[M,N] = (relu?) A[M,K] @ Bw[N,K]^T ; A,Bw bf16, C f32 or bf16.
// Block tile 256x128, 8 wave32 waves (4x2), 64x64 per wave:
// 16 v_wmma per K-step against 16 b128 loads (1.0 wmma/load).
__global__ __launch_bounds__(256) void gemm_bf16_kernel(
    const unsigned short* __restrict__ A, int lda,
    const unsigned short* __restrict__ Bw, int ldb,
    float* __restrict__ Cf, unsigned short* __restrict__ Cbf, int ldc,
    int K, int relu) {
  const int lane = threadIdx.x & 31;
  const int wave = threadIdx.x >> 5;
  const int wr = wave & 3, wc = wave >> 2;
  const int m0 = blockIdx.x * 256 + wr * 64;
  const int n0 = blockIdx.y * 128 + wc * 64;

  v8f acc[4][4] = {};
  for (int k0 = 0; k0 < K; k0 += 32) {
    if (k0 + 32 < K)
      __builtin_prefetch(A + (long)(m0 + (lane & 15)) * lda + k0 + 32, 0, 0);
    v16bf a[4], b[4];
#pragma unroll
    for (int i = 0; i < 4; ++i) a[i] = load_fragA(A, lda, m0 + i * 16, k0, lane);
#pragma unroll
    for (int j = 0; j < 4; ++j) b[j] = load_fragB(Bw, ldb, n0 + j * 16, k0, lane);
#pragma unroll
    for (int i = 0; i < 4; ++i)
#pragma unroll
      for (int j = 0; j < 4; ++j)
        acc[i][j] = WMMA_BF16(a[i], b[j], acc[i][j]);
  }

  const int colb = lane & 15;
  const int rsub = (lane >> 4) << 3;
#pragma unroll
  for (int im = 0; im < 4; ++im) {
#pragma unroll
    for (int jn = 0; jn < 4; ++jn) {
      int  col  = n0 + jn * 16 + colb;
      long rowb = (long)m0 + im * 16 + rsub;
#pragma unroll
      for (int j = 0; j < 8; ++j) {
        float v = acc[im][jn][j];
        if (relu) v = v > 0.f ? v : 0.f;
        if (Cbf) Cbf[(rowb + j) * ldc + col] = f2bf(v);
        else     Cf [(rowb + j) * ldc + col] = v;
      }
    }
  }
}

// ----------------------------- fused feature net ----------------------------
// feat = relu(relu(Xpad @ W1p^T) @ W2^T), layer-1 tile staged in LDS.
// Block: 32 rows of X, 256 output cols (blockIdx.y in {0,1}); 8 waves.
// X-tile staged via GLOBAL_LOAD_ASYNC_TO_LDS (ASYNCcnt) when available.
__global__ __launch_bounds__(256) void feat_fused_kernel(
    const unsigned short* __restrict__ Xpad,  // [BS, 64]
    const unsigned short* __restrict__ W1p,   // [H, 64]
    const unsigned short* __restrict__ W2b,   // [H, H]
    unsigned short* __restrict__ feat) {      // [BS, H]
  __shared__ unsigned short sX[32 * FPAD];    // 4 KB
  __shared__ unsigned short sT[32 * H_];      // 32 KB
  const int  lane = threadIdx.x & 31;
  const int  wave = threadIdx.x >> 5;
  const long mblk = (long)blockIdx.x * 32;
  const int  nblk = blockIdx.y * 256;

  // cooperative load of the 32x64 x-tile (16 bytes / thread)
#if __has_builtin(__builtin_amdgcn_global_load_async_to_lds_b128)
  __builtin_amdgcn_global_load_async_to_lds_b128(
      (__attribute__((address_space(1))) async_v4i*)(Xpad + mblk * FPAD + (size_t)threadIdx.x * 8),
      (__attribute__((address_space(3))) async_v4i*)(sX + threadIdx.x * 8),
      0, 0);
#if __has_builtin(__builtin_amdgcn_s_wait_asynccnt)
  __builtin_amdgcn_s_wait_asynccnt(0);
#else
  asm volatile("s_wait_asynccnt 0" ::: "memory");
#endif
#else
  *(u32x4*)(sX + threadIdx.x * 8) = *(const u32x4*)(Xpad + mblk * FPAD + threadIdx.x * 8);
#endif
  __syncthreads();

  // Stage 1: wave computes T1[0:32, wave*64 .. +64), K = 64
  {
    const int wn0 = wave * 64;
    v8f acc[2][4] = {};
#pragma unroll
    for (int k0 = 0; k0 < FPAD; k0 += 32) {
      v16bf a0 = load_fragA(sX, FPAD, 0,  k0, lane);
      v16bf a1 = load_fragA(sX, FPAD, 16, k0, lane);
      v16bf b[4];
#pragma unroll
      for (int j = 0; j < 4; ++j) b[j] = load_fragB(W1p, FPAD, wn0 + j * 16, k0, lane);
#pragma unroll
      for (int j = 0; j < 4; ++j) {
        acc[0][j] = WMMA_BF16(a0, b[j], acc[0][j]);
        acc[1][j] = WMMA_BF16(a1, b[j], acc[1][j]);
      }
    }
    const int colb = lane & 15, rsub = (lane >> 4) << 3;
#pragma unroll
    for (int im = 0; im < 2; ++im)
#pragma unroll
      for (int jn = 0; jn < 4; ++jn) {
        int col = wn0 + jn * 16 + colb;
        int r0  = im * 16 + rsub;
#pragma unroll
        for (int j = 0; j < 8; ++j) {
          float v = acc[im][jn][j];
          sT[(r0 + j) * H_ + col] = f2bf(v > 0.f ? v : 0.f);
        }
      }
  }
  __syncthreads();

  // Stage 2: wave computes feat[mblk..+32, nblk + wave*32 .. +32), K = 512
  {
    const int n0 = nblk + wave * 32;
    v8f acc[2][2] = {};
    for (int k0 = 0; k0 < H_; k0 += 32) {
      v16bf a0 = load_fragA(sT, H_, 0,  k0, lane);
      v16bf a1 = load_fragA(sT, H_, 16, k0, lane);
      v16bf b0 = load_fragB(W2b, H_, n0,      k0, lane);
      v16bf b1 = load_fragB(W2b, H_, n0 + 16, k0, lane);
      acc[0][0] = WMMA_BF16(a0, b0, acc[0][0]);
      acc[0][1] = WMMA_BF16(a0, b1, acc[0][1]);
      acc[1][0] = WMMA_BF16(a1, b0, acc[1][0]);
      acc[1][1] = WMMA_BF16(a1, b1, acc[1][1]);
    }
    const int colb = lane & 15, rsub = (lane >> 4) << 3;
#pragma unroll
    for (int im = 0; im < 2; ++im)
#pragma unroll
      for (int jn = 0; jn < 2; ++jn) {
        int  col  = n0 + jn * 16 + colb;
        long rowb = mblk + im * 16 + rsub;
#pragma unroll
        for (int j = 0; j < 8; ++j) {
          float v = acc[im][jn][j];
          feat[(rowb + j) * H_ + col] = f2bf(v > 0.f ? v : 0.f);
        }
      }
  }
}

// ----------------------------- GRU pointwise --------------------------------
// Layer-0 cell: h1 = n + z*(h - n) with gates from gi0 and gh[:,0:1536]
__global__ void gru_cell1_kernel(const float* __restrict__ gi,
                                 const float* __restrict__ gh,
                                 const float* __restrict__ h,
                                 unsigned short* __restrict__ h1bf) {
  int idx = blockIdx.x * blockDim.x + threadIdx.x;  // < B*H
  int b = idx >> 9, j = idx & (H_ - 1);
  const float* gib = gi + (size_t)b * 3 * H_;
  const float* ghb = gh + (size_t)b * 6 * H_;
  float r  = sigmoidf_(gib[j] + ghb[j]);
  float z  = sigmoidf_(gib[H_ + j] + ghb[H_ + j]);
  float n  = tanhf(gib[2 * H_ + j] + r * ghb[2 * H_ + j]);
  float hv = h[idx];
  h1bf[idx] = f2bf(n + z * (hv - n));
}

// Layer-1 cell + silence select: h = active ? (n + z*(h-n)) : h
__global__ void gru_cell2_kernel(const float* __restrict__ gi1,
                                 const float* __restrict__ gh,
                                 float* __restrict__ h,
                                 unsigned short* __restrict__ hbf,
                                 const int* __restrict__ flags, int s) {
  int idx = blockIdx.x * blockDim.x + threadIdx.x;
  int b = idx >> 9, j = idx & (H_ - 1);
  const float* gib = gi1 + (size_t)b * 3 * H_;
  const float* g2  = gh  + (size_t)b * 6 * H_ + 3 * H_;  // h @ Whh1^T block
  float hv = h[idx];
  float r  = sigmoidf_(gib[j] + g2[j]);
  float z  = sigmoidf_(gib[H_ + j] + g2[H_ + j]);
  float n  = tanhf(gib[2 * H_ + j] + r * g2[2 * H_ + j]);
  float h2 = n + z * (hv - n);
  float hn = flags[s] ? h2 : hv;
  h[idx]   = hn;
  hbf[idx] = f2bf(hn);
}

// ----------------------------- classifier head ------------------------------
// out[b,k] = sum_j c1[b,j] * Wc2[k,j]   (c1 bf16 [B,256], Wc2 f32 [10,256])
__global__ void head_kernel(const unsigned short* __restrict__ c1,
                            const float* __restrict__ Wc2,
                            float* __restrict__ out) {
  int idx = blockIdx.x * blockDim.x + threadIdx.x;
  if (idx >= B_ * K_) return;
  int b = idx / K_, k = idx % K_;
  float s = 0.f;
#pragma unroll 8
  for (int j = 0; j < H_ / 2; ++j)
    s += bf2f(c1[b * (H_ / 2) + j]) * Wc2[k * (H_ / 2) + j];
  out[idx] = s;
}

// ----------------------------- launch --------------------------------------

extern "C" void kernel_launch(void* const* d_in, const int* in_sizes, int n_in,
                              void* d_out, int out_size, void* d_ws, size_t ws_size,
                              hipStream_t stream) {
  (void)in_sizes; (void)n_in; (void)out_size; (void)ws_size;
  const float* x    = (const float*)d_in[0];
  const float* W1   = (const float*)d_in[1];
  const float* W2   = (const float*)d_in[2];
  const float* Wih0 = (const float*)d_in[3];
  const float* Whh0 = (const float*)d_in[4];
  const float* Wih1 = (const float*)d_in[5];
  const float* Whh1 = (const float*)d_in[6];
  const float* Wc1  = (const float*)d_in[7];
  const float* Wc2  = (const float*)d_in[8];
  float* out = (float*)d_out;
  char*  ws  = (char*)d_ws;

  size_t off = 0;
  auto alloc = [&](size_t bytes) -> size_t {
    size_t o = off;
    off += (bytes + 255) & ~(size_t)255;
    return o;
  };
  size_t o_flags = alloc(S_ * sizeof(int));
  size_t o_h     = alloc((size_t)B_ * H_ * 4);
  size_t o_hbf   = alloc((size_t)B_ * H_ * 2);
  size_t o_h1bf  = alloc((size_t)B_ * H_ * 2);
  size_t zeroBytes = off;                          // zero everything above
  size_t o_xpad  = alloc((size_t)BS * FPAD * 2);
  size_t o_w1p   = alloc((size_t)H_ * FPAD * 2);
  size_t o_w2b   = alloc((size_t)H_ * H_ * 2);
  size_t o_wih0  = alloc((size_t)3 * H_ * H_ * 2);
  size_t o_whhc  = alloc((size_t)6 * H_ * H_ * 2); // [Whh0 ; Whh1] concat
  size_t o_wih1  = alloc((size_t)3 * H_ * H_ * 2);
  size_t o_wc1   = alloc((size_t)(H_ / 2) * H_ * 2);
  size_t o_feat  = alloc((size_t)BS * H_ * 2);
  size_t o_gi0   = alloc((size_t)B_ * 3 * H_ * 4);
  size_t o_gh    = alloc((size_t)B_ * 6 * H_ * 4);
  size_t o_gi1   = alloc((size_t)B_ * 3 * H_ * 4);
  size_t o_c1    = alloc((size_t)B_ * (H_ / 2) * 2);

  int*            flags = (int*)(ws + o_flags);
  float*          h     = (float*)(ws + o_h);
  unsigned short* hbf   = (unsigned short*)(ws + o_hbf);
  unsigned short* h1bf  = (unsigned short*)(ws + o_h1bf);
  unsigned short* xpad  = (unsigned short*)(ws + o_xpad);
  unsigned short* w1p   = (unsigned short*)(ws + o_w1p);
  unsigned short* w2b   = (unsigned short*)(ws + o_w2b);
  unsigned short* wih0b = (unsigned short*)(ws + o_wih0);
  unsigned short* whhc  = (unsigned short*)(ws + o_whhc);
  unsigned short* wih1b = (unsigned short*)(ws + o_wih1);
  unsigned short* wc1b  = (unsigned short*)(ws + o_wc1);
  unsigned short* feat  = (unsigned short*)(ws + o_feat);
  float*          gi0   = (float*)(ws + o_gi0);
  float*          gh    = (float*)(ws + o_gh);
  float*          gi1   = (float*)(ws + o_gi1);
  unsigned short* c1    = (unsigned short*)(ws + o_c1);

  // zero flags + hidden state
  {
    long n = (long)(zeroBytes / 4);
    zero_u32_kernel<<<(unsigned)((n + 255) / 256), 256, 0, stream>>>((unsigned int*)ws, n);
  }
  energy_kernel<<<S_, 256, 0, stream>>>(x, flags);

  // bf16 conversions (with K padding 40 -> 64 for x and W1)
  auto cvt = [&](const float* s, unsigned short* d, int rows, int kin, int kout) {
    long n = (long)rows * kout;
    convert_pad_kernel<<<(unsigned)((n + 255) / 256), 256, 0, stream>>>(s, d, rows, kin, kout);
  };
  cvt(x,    xpad,  BS,      F_, FPAD);
  cvt(W1,   w1p,   H_,      F_, FPAD);
  cvt(W2,   w2b,   H_,      H_, H_);
  cvt(Wih0, wih0b, 3 * H_,  H_, H_);
  cvt(Whh0, whhc,               3 * H_, H_, H_);
  cvt(Whh1, whhc + (size_t)3 * H_ * H_, 3 * H_, H_, H_);
  cvt(Wih1, wih1b, 3 * H_,  H_, H_);
  cvt(Wc1,  wc1b,  H_ / 2,  H_, H_);

  // feature net (fused two-layer)
  feat_fused_kernel<<<dim3(BS / 32, 2), 256, 0, stream>>>(xpad, w1p, w2b, feat);

  // recurrent scan: 128 sequential steps
  for (int s = 0; s < S_; ++s) {
    // gi0 = feat[:, s, :] @ Wih0^T                 [512 x 1536]
    gemm_bf16_kernel<<<dim3(B_ / 256, (3 * H_) / 128), 256, 0, stream>>>(
        feat + (size_t)s * H_, S_ * H_, wih0b, H_, gi0, nullptr, 3 * H_, H_, 0);
    // gh = h @ [Whh0;Whh1]^T                       [512 x 3072]
    gemm_bf16_kernel<<<dim3(B_ / 256, (6 * H_) / 128), 256, 0, stream>>>(
        hbf, H_, whhc, H_, gh, nullptr, 6 * H_, H_, 0);
    gru_cell1_kernel<<<(B_ * H_) / 256, 256, 0, stream>>>(gi0, gh, h, h1bf);
    // gi1 = h1 @ Wih1^T                            [512 x 1536]
    gemm_bf16_kernel<<<dim3(B_ / 256, (3 * H_) / 128), 256, 0, stream>>>(
        h1bf, H_, wih1b, H_, gi1, nullptr, 3 * H_, H_, 0);
    gru_cell2_kernel<<<(B_ * H_) / 256, 256, 0, stream>>>(gi1, gh, h, hbf, flags, s);
  }

  // head: c1 = relu(h @ Wc1^T) in bf16, then small dense to [B, 10]
  gemm_bf16_kernel<<<dim3(B_ / 256, (H_ / 2) / 128), 256, 0, stream>>>(
      hbf, H_, wc1b, H_, nullptr, c1, H_ / 2, H_, 1);
  head_kernel<<<(B_ * K_ + 255) / 256, 256, 0, stream>>>(c1, Wc2, out);
}
